// RelationalAttentionLayer_20959440405253
// MI455X (gfx1250) — matrix operand order repaired and checked by reading, not compiled, not run on hardware
//
#include <hip/hip_runtime.h>
#include <hip/hip_bf16.h>

typedef __attribute__((ext_vector_type(16))) _Float16 v16h;
typedef __attribute__((ext_vector_type(8)))  float    v8f;

#define NEGV (-1000000000.0f)

// ---------------------------------------------------------------------------
// A-fragment loader (16x32 f16, row-major source):
// lane m = lane&15, g = lane>>4; e<8 -> K = k0+8g+e ; e>=8 -> K = k0+16+8g+(e-8)
// ---------------------------------------------------------------------------
__device__ inline v16h load_a_f16(const _Float16* __restrict__ row, int k0, int g) {
    v16h a = {};
    const _Float16* p1 = row + k0 + 8 * g;
    const _Float16* p2 = p1 + 16;
#pragma unroll
    for (int e = 0; e < 8; ++e) { a[e] = p1[e]; a[8 + e] = p2[e]; }
    return a;
}

// B-fragment (32x16 f16): lane holds column N = c0+(lane&15); e<8 -> K = k0+8g+e ; e>=8 -> K = k0+16+8g+(e-8)
__device__ inline v16h load_b_f16(const _Float16* __restrict__ B, int ldb, int k0,
                                  int lane, int c0) {
    v16h b = {};
    const _Float16* p = B + (size_t)(k0 + 8 * (lane >> 4)) * ldb + c0 + (lane & 15);
#pragma unroll
    for (int e = 0; e < 8; ++e) { b[e] = p[(size_t)e * ldb]; b[8 + e] = p[(size_t)(16 + e) * ldb]; }
    return b;
}

// ---------------------------------------------------------------------------
// Kernel 0: convert the three weight matrices (each 32768 f32) to f16
// ---------------------------------------------------------------------------
__global__ void cvt_weights_kernel(const float* __restrict__ Wn,
                                   const float* __restrict__ W1,
                                   const float* __restrict__ W2,
                                   _Float16* __restrict__ Wn_h,
                                   _Float16* __restrict__ W1_h,
                                   _Float16* __restrict__ W2_h) {
    int i = blockIdx.x * 256 + threadIdx.x;
    if (i < 32768) {
        Wn_h[i] = (_Float16)Wn[i];
        W1_h[i] = (_Float16)W1[i];
        W2_h[i] = (_Float16)W2[i];
    }
}

// ---------------------------------------------------------------------------
// Kernel 1: h = visit_matrix @ W_node   (32000x256)@(256x128)
// grid = 2000 (16 rows each), block = 256 (8 waves, one 16-col tile per wave)
// A tile staged once per block in LDS as f16 (kills the 8x redundant A loads)
// ---------------------------------------------------------------------------
__global__ void proj_kernel(const float* __restrict__ X,
                            const _Float16* __restrict__ Wn_h,
                            float* __restrict__ h) {
    __shared__ _Float16 s_a[16 * 256];  // 8 KB
    const int tid = threadIdx.x;
    const size_t m0 = (size_t)blockIdx.x * 16;

    for (int i = tid; i < 4096; i += 256) s_a[i] = (_Float16)X[m0 * 256 + i];
    __syncthreads();

    const int wave = tid >> 5;
    const int lane = tid & 31;
    const int g = lane >> 4, l15 = lane & 15;
    const int c0 = wave * 16;

    v8f acc = {};
#pragma unroll
    for (int k0 = 0; k0 < 256; k0 += 32) {
        v16h a = load_a_f16(s_a + l15 * 256, k0, g);
        v16h b = load_b_f16(Wn_h, 128, k0, lane, c0);
        acc = __builtin_amdgcn_wmma_f32_16x16x32_f16(false, a, false, b,
                                                     (short)0, acc, false, false);
    }
#pragma unroll
    for (int r = 0; r < 8; ++r)
        h[(m0 + r + 8 * g) * 128 + c0 + l15] = acc[r];
}

// ---------------------------------------------------------------------------
// Kernel 2: fused per-node relational attention (+ LN1) -> h_upd
// grid = 2000 (one node per block), block = 256, dynamic LDS ~150 KB
// ---------------------------------------------------------------------------
__global__ void attn_kernel(const float* __restrict__ h,
                            const int* __restrict__ visit_days,
                            const int* __restrict__ neighbor_idx,
                            const int* __restrict__ rel_type,
                            const float* __restrict__ edge_vec,
                            const float* __restrict__ attn_w,
                            const float* __restrict__ ln1_g,
                            const float* __restrict__ ln1_b,
                            float* __restrict__ h_upd) {
    extern __shared__ char smem[];
    float*    s_agg  = (float*)smem;                 // [t][k][128]  131072 B
    float*    s_h    = s_agg + 16 * 16 * 128;        // [t][128]       8192 B
    _Float16* s_tw   = (_Float16*)(s_h + 16 * 128);  // [k][t][16]     8192 B
    float*    s_score = (float*)(s_tw + 16 * 16 * 16); // [t*16+k]     1024 B
    float*    s_wk   = s_score + 256;                // [t*16+k]       1024 B
    float*    s_va   = s_wk + 256;                   // [t*16+k]       1024 B
    float*    s_red  = s_va + 256;                   //                1024 B
    int*      s_days = (int*)(s_red + 256);          // 16
    int*      s_nb   = s_days + 16;                  // 16
    int*      s_rel  = s_nb + 16;                    // 16
    int*      s_nbd  = s_rel + 16;                   // [k][s] 256
    float*    s_hdot = (float*)(s_nbd + 256);        // 16
    float*    s_edot = s_hdot + 16;                  // 16
    float*    s_mu   = s_edot + 16;                  // 16
    float*    s_rsig = s_mu + 16;                    // 16
    float*    s_np   = s_rsig + 16;                  // 1

    const int tid = threadIdx.x;
    const int n = blockIdx.x;

    // phase 1: async-copy own h rows (8192 B) into LDS; scalars in parallel.
    // ASYNCcnt tracks the copies; flat->LDS address = low 32 bits of pointer.
    {
        unsigned ldsoff = (unsigned)(uintptr_t)s_h + (unsigned)tid * 16u;
        unsigned long long gaddr =
            (unsigned long long)(uintptr_t)(h + (size_t)n * 2048) +
            (unsigned long long)tid * 16ull;
        asm volatile("global_load_async_to_lds_b128 %0, %1, off"
                     :: "v"(ldsoff), "v"(gaddr) : "memory");
        asm volatile("global_load_async_to_lds_b128 %0, %1, off offset:4096"
                     :: "v"(ldsoff), "v"(gaddr) : "memory");
    }
    if (tid < 16) {
        s_days[tid] = visit_days[n * 16 + tid];
        s_nb[tid]   = neighbor_idx[n * 16 + tid];
        s_rel[tid]  = rel_type[n * 16 + tid];
    }
    asm volatile("s_wait_asynccnt 0x0" ::: "memory");
    __syncthreads();

    // phase 2: prefetch neighbor rows, neighbor days, zero wk, hdot/edot/np
    // 16 neighbors x 2048 f32 = 1024 cachelines of 128B -> 4 prefetches/thread
    for (int i = tid; i < 16 * 64; i += 256) {
        int k = i >> 6;
        int off = (i & 63) * 32;
        __builtin_prefetch(h + (size_t)s_nb[k] * 2048 + off, 0, 1);
    }
    {
        int k = tid >> 4, s = tid & 15;
        s_nbd[tid] = visit_days[s_nb[k] * 16 + s];
    }
    s_wk[tid] = 0.0f;
    if (tid < 16) {
        float acc = 0.f;
        for (int o = 0; o < 128; ++o) acc += s_h[tid * 128 + o] * attn_w[o];
        s_hdot[tid] = acc;
    } else if (tid < 32) {
        int k = tid - 16;
        float acc = 0.f;
        const float* ev = edge_vec + ((size_t)n * 16 + k) * 32;
        for (int o = 0; o < 32; ++o) acc += ev[o] * attn_w[128 + o];
        s_edot[k] = acc;
    } else if (tid == 32) {
        float np = 0.f;
        for (int r = 0; r < 4; ++r) {
            int p = 0;
            for (int k = 0; k < 16; ++k) p |= (s_rel[k] == r);
            np += (float)p;
        }
        s_np[0] = fmaxf(np, 1.0f);
    }
    __syncthreads();

    // phase 3: temporal softmax tw[t][k][s] over s  (one thread per (t,k))
    {
        int t = tid >> 4, k = tid & 15;
        float sc[16];
        float mx = NEGV;
        int anyv = 0;
        int dt = s_days[t];
#pragma unroll
        for (int s = 0; s < 16; ++s) {
            int diff = dt - s_nbd[k * 16 + s];
            bool v = diff > 0;
            float x = v ? 1.0f / (1.0f + (float)diff) : NEGV;
            sc[s] = x;
            mx = fmaxf(mx, x);
            anyv |= (int)v;
        }
        float sum = 0.f;
#pragma unroll
        for (int s = 0; s < 16; ++s) { float e = __expf(sc[s] - mx); sc[s] = e; sum += e; }
        float inv = 1.0f / sum;
#pragma unroll
        for (int s = 0; s < 16; ++s) s_tw[(k * 16 + t) * 16 + s] = (_Float16)(sc[s] * inv);
        s_va[t * 16 + k] = anyv ? 1.0f : 0.0f;
    }
    __syncthreads();

    // phase 4: agg[t][k][:] = tw[t][k][:] @ h[nb[k]]  via WMMA (K=16 zero-padded)
    {
        const int lane = tid & 31, g = lane >> 4, l15 = lane & 15;
        const int c0 = (tid >> 5) * 16;
        for (int k = 0; k < 16; ++k) {
            v16h a = {};
            const _Float16* pa = s_tw + (k * 16 + l15) * 16 + 8 * g;
#pragma unroll
            for (int e = 0; e < 8; ++e) a[e] = pa[e];   // K>=16 stays zero
            v16h b = {};
            {                                           // K rows 16..31 stay zero
                const float* pb = h + ((size_t)s_nb[k] * 16 + 8 * g) * 128 + c0 + l15;
#pragma unroll
                for (int e = 0; e < 8; ++e) b[e] = (_Float16)pb[e * 128];
            }
            v8f c = {};
            c = __builtin_amdgcn_wmma_f32_16x16x32_f16(false, a, false, b,
                                                       (short)0, c, false, false);
#pragma unroll
            for (int r = 0; r < 8; ++r) {
                int t = r + 8 * g;
                s_agg[(t * 16 + k) * 128 + c0 + l15] = c[r] * s_va[t * 16 + k];
            }
        }
    }
    __syncthreads();

    // phase 5: scores + leaky relu (one thread per (t,k))
    {
        int t = tid >> 4, k = tid & 15;
        float acc = 0.f;
        const float* pa = s_agg + (t * 16 + k) * 128;
        for (int o = 0; o < 128; ++o) acc += pa[o] * attn_w[160 + o];
        float sc = s_hdot[t] + s_edot[k] + acc;
        s_score[t * 16 + k] = sc > 0.f ? sc : 0.2f * sc;
    }
    __syncthreads();

    // phase 6: per-(t,r) softmax over k restricted to rel_type==r
    if (tid < 64) {
        int t = tid >> 2, r = tid & 3;
        float mx = NEGV;
        for (int k = 0; k < 16; ++k)
            if (s_rel[k] == r) mx = fmaxf(mx, s_score[t * 16 + k]);
        if (mx > 0.5f * NEGV) {  // relation present
            float e[16];
            float sum = 0.f;
            for (int k = 0; k < 16; ++k) {
                e[k] = (s_rel[k] == r) ? __expf(s_score[t * 16 + k] - mx) : 0.f;
                sum += e[k];
            }
            float inv = 1.0f / sum;
            for (int k = 0; k < 16; ++k)
                if (s_rel[k] == r) s_wk[t * 16 + k] = e[k] * inv;
        }
    }
    __syncthreads();

    // phase 7: combined + residual + LayerNorm1 -> h_upd
    {
        int t = tid >> 4, cg = tid & 15;  // 8 columns per thread
        float cb[8];
#pragma unroll
        for (int j = 0; j < 8; ++j) cb[j] = 0.f;
        for (int k = 0; k < 16; ++k) {
            float w = s_wk[t * 16 + k];
            if (w != 0.f) {
                const float* pa = s_agg + (t * 16 + k) * 128 + cg * 8;
#pragma unroll
                for (int j = 0; j < 8; ++j) cb[j] += w * pa[j];
            }
        }
        float invnp = 1.0f / s_np[0];
        float psum = 0.f;
#pragma unroll
        for (int j = 0; j < 8; ++j) {
            cb[j] = s_h[t * 128 + cg * 8 + j] + cb[j] * invnp;
            psum += cb[j];
        }
        s_red[tid] = psum;
        __syncthreads();
        if (cg == 0) {
            float m = 0.f;
            for (int i = 0; i < 16; ++i) m += s_red[t * 16 + i];
            s_mu[t] = m * (1.0f / 128.0f);
        }
        __syncthreads();
        float mu = s_mu[t], pv = 0.f;
#pragma unroll
        for (int j = 0; j < 8; ++j) { float d = cb[j] - mu; pv += d * d; }
        s_red[tid] = pv;
        __syncthreads();
        if (cg == 0) {
            float v = 0.f;
            for (int i = 0; i < 16; ++i) v += s_red[t * 16 + i];
            s_rsig[t] = rsqrtf(v * (1.0f / 128.0f) + 1e-5f);
        }
        __syncthreads();
        float rs = s_rsig[t];
#pragma unroll
        for (int j = 0; j < 8; ++j) {
            int o = cg * 8 + j;
            h_upd[(size_t)n * 2048 + t * 128 + o] = (cb[j] - mu) * rs * ln1_g[o] + ln1_b[o];
        }
    }
}

// ---------------------------------------------------------------------------
// Kernel 3: FFN (128->256 relu ->128) + residual + LN2 -> out
// grid = 2000 (16 rows each), block = 256 (8 waves)
// ---------------------------------------------------------------------------
__global__ void ffn_kernel(const float* __restrict__ h_upd,
                           const _Float16* __restrict__ W1_h,
                           const float* __restrict__ b1,
                           const _Float16* __restrict__ W2_h,
                           const float* __restrict__ b2,
                           const float* __restrict__ ln2_g,
                           const float* __restrict__ ln2_b,
                           float* __restrict__ out) {
    __shared__ float    s_hu[16 * 128];
    __shared__ _Float16 s_hu_h[16 * 128];
    __shared__ _Float16 s_mid[16 * 256];
    __shared__ float    s_out[16 * 128];
    __shared__ float    s_red[256];
    __shared__ float    s_mu[16], s_rsig[16];

    const int tid = threadIdx.x;
    const size_t m0 = (size_t)blockIdx.x * 16;
    for (int i = tid; i < 2048; i += 256) {
        float v = h_upd[m0 * 128 + i];
        s_hu[i] = v;
        s_hu_h[i] = (_Float16)v;
    }
    __syncthreads();

    const int lane = tid & 31, g = lane >> 4, l15 = lane & 15, wave = tid >> 5;

    // GEMM1: (16x128)@(128x256) + b1, ReLU -> s_mid (f16)
#pragma unroll
    for (int ti = 0; ti < 2; ++ti) {
        int c0 = (wave + ti * 8) * 16;
        v8f acc = {};
#pragma unroll
        for (int k0 = 0; k0 < 128; k0 += 32) {
            v16h a = load_a_f16(s_hu_h + l15 * 128, k0, g);
            v16h b = load_b_f16(W1_h, 256, k0, lane, c0);
            acc = __builtin_amdgcn_wmma_f32_16x16x32_f16(false, a, false, b,
                                                         (short)0, acc, false, false);
        }
#pragma unroll
        for (int r = 0; r < 8; ++r) {
            int t = r + 8 * g, col = c0 + l15;
            float v = acc[r] + b1[col];
            s_mid[t * 256 + col] = (_Float16)(v > 0.f ? v : 0.f);
        }
    }
    __syncthreads();

    // GEMM2: (16x256)@(256x128) + b2 + residual -> s_out (f32)
    {
        int c0 = wave * 16;
        v8f acc = {};
#pragma unroll
        for (int k0 = 0; k0 < 256; k0 += 32) {
            v16h a = load_a_f16(s_mid + l15 * 256, k0, g);
            v16h b = load_b_f16(W2_h, 128, k0, lane, c0);
            acc = __builtin_amdgcn_wmma_f32_16x16x32_f16(false, a, false, b,
                                                         (short)0, acc, false, false);
        }
#pragma unroll
        for (int r = 0; r < 8; ++r) {
            int t = r + 8 * g, col = c0 + l15;
            s_out[t * 128 + col] = acc[r] + b2[col] + s_hu[t * 128 + col];
        }
    }
    __syncthreads();

    // LN2
    {
        int t = tid >> 4, cg = tid & 15;
        float x[8];
        float psum = 0.f;
#pragma unroll
        for (int j = 0; j < 8; ++j) { x[j] = s_out[t * 128 + cg * 8 + j]; psum += x[j]; }
        s_red[tid] = psum;
        __syncthreads();
        if (cg == 0) {
            float m = 0.f;
            for (int i = 0; i < 16; ++i) m += s_red[t * 16 + i];
            s_mu[t] = m * (1.0f / 128.0f);
        }
        __syncthreads();
        float mu = s_mu[t], pv = 0.f;
#pragma unroll
        for (int j = 0; j < 8; ++j) { float d = x[j] - mu; pv += d * d; }
        s_red[tid] = pv;
        __syncthreads();
        if (cg == 0) {
            float v = 0.f;
            for (int i = 0; i < 16; ++i) v += s_red[t * 16 + i];
            s_rsig[t] = rsqrtf(v * (1.0f / 128.0f) + 1e-5f);
        }
        __syncthreads();
        float rs = s_rsig[t];
#pragma unroll
        for (int j = 0; j < 8; ++j) {
            int o = cg * 8 + j;
            out[m0 * 128 + t * 128 + o] = (x[j] - mu) * rs * ln2_g[o] + ln2_b[o];
        }
    }
}

// ---------------------------------------------------------------------------
extern "C" void kernel_launch(void* const* d_in, const int* in_sizes, int n_in,
                              void* d_out, int out_size, void* d_ws, size_t ws_size,
                              hipStream_t stream) {
    const float* visit_matrix = (const float*)d_in[0];
    const int*   visit_days   = (const int*)d_in[1];
    const int*   neighbor_idx = (const int*)d_in[2];
    const int*   rel_type     = (const int*)d_in[3];
    const float* edge_vec     = (const float*)d_in[4];
    const float* W_node       = (const float*)d_in[5];
    const float* attn_w       = (const float*)d_in[6];
    const float* ln1_g        = (const float*)d_in[7];
    const float* ln1_b        = (const float*)d_in[8];
    const float* W1           = (const float*)d_in[9];
    const float* b1           = (const float*)d_in[10];
    const float* W2           = (const float*)d_in[11];
    const float* b2           = (const float*)d_in[12];
    const float* ln2_g        = (const float*)d_in[13];
    const float* ln2_b        = (const float*)d_in[14];
    float* out = (float*)d_out;

    // workspace carve-up
    char* ws = (char*)d_ws;
    float* h      = (float*)ws;                         // 32000*128*4 = 16,384,000 B
    float* h_upd  = (float*)(ws + 16384000);            // 16,384,000 B
    _Float16* Wn_h = (_Float16*)(ws + 32768000);        // 65,536 B
    _Float16* W1_h = Wn_h + 32768;                      // 65,536 B
    _Float16* W2_h = W1_h + 32768;                      // 65,536 B

    cvt_weights_kernel<<<128, 256, 0, stream>>>(W_node, W1, W2, Wn_h, W1_h, W2_h);

    proj_kernel<<<2000, 256, 0, stream>>>(visit_matrix, Wn_h, h);

    // dynamic LDS: agg(131072)+h(8192)+tw(8192)+4*256f(4096)+ints(192+1024)+misc(260)
    size_t lds2 = 131072 + 8192 + 8192 + 4096 + 192 + 1024 + 512;
    attn_kernel<<<2000, 256, lds2, stream>>>(h, visit_days, neighbor_idx, rel_type,
                                             edge_vec, attn_w, ln1_g, ln1_b, h_upd);

    ffn_kernel<<<2000, 256, 0, stream>>>(h_upd, W1_h, b1, W2_h, b2, ln2_g, ln2_b, out);
}
